// AnchorAttention_66022237274174
// MI455X (gfx1250) — matrix operand order repaired and checked by reading, not compiled, not run on hardware
//
#include <hip/hip_runtime.h>
#include <hip/hip_bf16.h>

typedef __attribute__((ext_vector_type(16))) __bf16 v16bf;
typedef __attribute__((ext_vector_type(8)))  float  v8f;
typedef unsigned int uint32;

constexpr int B_ = 4, N_ = 4096, D_ = 1024, H_ = 16, HD_ = 64, KA_ = 256;
constexpr float SCALE_ = 0.125f;   // 1/sqrt(64)

// ---------------------------------------------------------------- convert (f32 -> bf16, x4 vectorized)
__global__ void __launch_bounds__(256) cvt_f32_bf16_k(const float* __restrict__ in,
                                                      __bf16* __restrict__ out, int n4) {
    int i = blockIdx.x * 256 + threadIdx.x;
    if (i < n4) {
        float4 f = ((const float4*)in)[i];
        union { __bf16 b[2]; uint32 u; } t0, t1;
        t0.b[0] = (__bf16)f.x; t0.b[1] = (__bf16)f.y;
        t1.b[0] = (__bf16)f.z; t1.b[1] = (__bf16)f.w;
        ((uint32*)out)[2 * i]     = t0.u;
        ((uint32*)out)[2 * i + 1] = t1.u;
    }
}

// ---------------------------------------------------------------- GEMM (bf16 in, WMMA f32 acc)
// C[M,N] = A[M,K] @ W[K,N] + bias
// block tile 128x128, 256 threads = 8 waves (4M x 2N), wave tile 32x64 (acc[2][4]),
// K-step 32, double-buffered LDS: one barrier per k-step, loads overlap WMMAs.
template <bool OUTF32>
__global__ void __launch_bounds__(256)
gemm_bf16_wmma_k(const __bf16* __restrict__ A, int lda, long sA,
                 const __bf16* __restrict__ W, int ldw,
                 const float* __restrict__ bias,
                 void* __restrict__ Cv, int ldc, long sC, int K)
{
    __shared__ __bf16 As[2][128][36];   // [buf][m][k]
    __shared__ __bf16 Bt[2][128][36];   // [buf][n][k] (transposed: frag pairs contiguous)

    const int tid  = threadIdx.x;
    const int lane = tid & 31, wave = tid >> 5;
    const int waveM = wave >> 1, waveN = wave & 1;
    const int khalf = lane >> 4;
    const int row0 = blockIdx.y * 128;
    const int n0   = blockIdx.x * 128;
    const __bf16* Ab = A + (long)blockIdx.z * sA;

    v8f acc[2][4] = {};

    auto loadTile = [&](int k0, int buf) {
        // A tile 128x32 : 2048 dwords, 8 per thread, coalesced over k-pairs
        #pragma unroll
        for (int i = 0; i < 8; ++i) {
            int idx = tid + 256 * i;
            int r = idx >> 4, c2 = idx & 15;
            uint32 u = *(const uint32*)(Ab + (long)(row0 + r) * lda + k0 + 2 * c2);
            *(uint32*)&As[buf][r][2 * c2] = u;
        }
        // W tile 32x128 : read coalesced over n, scatter transposed into Bt[n][k]
        #pragma unroll
        for (int i = 0; i < 8; ++i) {
            int idx = tid + 256 * i;
            int kk = idx >> 6, n2 = idx & 63;
            uint32 u = *(const uint32*)(W + (long)(k0 + kk) * ldw + n0 + 2 * n2);
            __bf16 b0 = ((const __bf16*)&u)[0], b1 = ((const __bf16*)&u)[1];
            Bt[buf][2 * n2][kk]     = b0;
            Bt[buf][2 * n2 + 1][kk] = b1;
        }
    };

    auto compute = [&](int buf) {
        v16bf af[2], bfr[4];
        #pragma unroll
        for (int mt = 0; mt < 2; ++mt) {
            int ar = waveM * 32 + mt * 16 + (lane & 15);
            #pragma unroll
            for (int vv = 0; vv < 8; ++vv) {
                int kd = (vv < 4) ? (khalf * 8 + 2 * vv) : (16 + khalf * 8 + 2 * (vv - 4));
                af[mt][2 * vv]     = As[buf][ar][kd];
                af[mt][2 * vv + 1] = As[buf][ar][kd + 1];
            }
        }
        #pragma unroll
        for (int nt = 0; nt < 4; ++nt) {
            int bc = waveN * 64 + nt * 16 + (lane & 15);
            #pragma unroll
            for (int vv = 0; vv < 8; ++vv) {
                int kd = (vv < 4) ? (khalf * 8 + 2 * vv) : (16 + khalf * 8 + 2 * (vv - 4));
                bfr[nt][2 * vv]     = Bt[buf][bc][kd];
                bfr[nt][2 * vv + 1] = Bt[buf][bc][kd + 1];
            }
        }
        #pragma unroll
        for (int mt = 0; mt < 2; ++mt)
            #pragma unroll
            for (int nt = 0; nt < 4; ++nt)
                acc[mt][nt] = __builtin_amdgcn_wmma_f32_16x16x32_bf16(
                    false, af[mt], false, bfr[nt], (short)0, acc[mt][nt], false, false);
    };

    loadTile(0, 0);
    __syncthreads();
    int buf = 0;
    for (int k0 = 32; k0 < K; k0 += 32) {
        loadTile(k0, buf ^ 1);   // global->LDS for next tile overlaps WMMAs below
        compute(buf);
        __syncthreads();
        buf ^= 1;
    }
    compute(buf);

    // epilogue (branch-free, templated output type)
    const long cbase = (long)blockIdx.z * sC;
    #pragma unroll
    for (int mt = 0; mt < 2; ++mt) {
        #pragma unroll
        for (int nt = 0; nt < 4; ++nt) {
            int col = n0 + waveN * 64 + nt * 16 + (lane & 15);
            float bb = bias[col];
            #pragma unroll
            for (int r = 0; r < 8; ++r) {
                int row = row0 + waveM * 32 + mt * 16 + r + 8 * khalf;
                float val = acc[mt][nt][r] + bb;
                if (OUTF32) ((float*)Cv)[cbase + (long)row * ldc + col] = val;
                else        ((__bf16*)Cv)[cbase + (long)row * ldc + col] = (__bf16)val;
            }
        }
    }
}

// ---------------------------------------------------------------- attention
// grid (N/64, H, B), block 128 (4 waves); each wave: 16 query rows x 256 keys x 64 dims
constexpr int KS_ELEMS  = KA_ * (HD_ + 8);   // K natural [key][d]
constexpr int VT_ELEMS  = HD_ * (KA_ + 8);   // V transposed [d][key]
constexpr int QS_ELEMS  = 4 * 16 * (HD_ + 8);
constexpr size_t SC_OFF = (size_t)(KS_ELEMS + VT_ELEMS + QS_ELEMS) * 2;
constexpr size_t ATTN_SMEM = SC_OFF + (size_t)64 * KA_ * 4 + 64 * 4;

__global__ void __launch_bounds__(128)
attn_wmma_k(const __bf16* __restrict__ q, const __bf16* __restrict__ kmat,
            const __bf16* __restrict__ vmat, __bf16* __restrict__ ctx)
{
    extern __shared__ char smem[];
    __bf16* kS   = (__bf16*)smem;          // [KA][HD+8]
    __bf16* vT   = kS + KS_ELEMS;          // [HD][KA+8]
    __bf16* qS   = vT + VT_ELEMS;          // [4][16][HD+8]
    float*  sc   = (float*)(smem + SC_OFF);// [4][16][KA]
    float*  rinv = sc + 64 * KA_;          // [64]

    const int tid = threadIdx.x, lane = tid & 31, wave = tid >> 5;
    const int h = blockIdx.y, b = blockIdx.z;
    const int qBlock = blockIdx.x * 64;
    const int khalf = lane >> 4;

    const __bf16* kB = kmat + ((long)b * KA_) * D_ + h * HD_;
    const __bf16* vB = vmat + ((long)b * KA_) * D_ + h * HD_;
    const __bf16* qB = q    + ((long)b * N_ + qBlock) * D_ + h * HD_;

    // stage K (natural) and V (transposed)
    for (int i = tid; i < KA_ * HD_ / 2; i += 128) {
        int kk = i >> 5, d2 = i & 31;
        uint32 uk = *(const uint32*)(kB + (long)kk * D_ + 2 * d2);
        uint32 uv = *(const uint32*)(vB + (long)kk * D_ + 2 * d2);
        *(uint32*)&kS[kk * (HD_ + 8) + 2 * d2] = uk;
        __bf16 v0 = ((const __bf16*)&uv)[0], v1 = ((const __bf16*)&uv)[1];
        vT[(2 * d2) * (KA_ + 8) + kk]     = v0;
        vT[(2 * d2 + 1) * (KA_ + 8) + kk] = v1;
    }
    // stage this wave's 16 q rows
    for (int i = lane; i < 16 * 32; i += 32) {
        int r = i >> 5, d2 = i & 31;
        uint32 u = *(const uint32*)(qB + (long)(wave * 16 + r) * D_ + 2 * d2);
        *(uint32*)&qS[(wave * 16 + r) * (HD_ + 8) + 2 * d2] = u;
    }
    __syncthreads();

    // q A-fragments for the two d-steps (reused across all 16 key tiles)
    v16bf aq[2];
    const __bf16* qrow = &qS[(wave * 16 + (lane & 15)) * (HD_ + 8)];
    #pragma unroll
    for (int s = 0; s < 2; ++s) {
        #pragma unroll
        for (int vv = 0; vv < 8; ++vv) {
            int kd = s * 32 + ((vv < 4) ? (khalf * 8 + 2 * vv) : (16 + khalf * 8 + 2 * (vv - 4)));
            aq[s][2 * vv]     = qrow[kd];
            aq[s][2 * vv + 1] = qrow[kd + 1];
        }
    }

    // scores = q @ k^T * scale  -> sc (f32)
    float* scw = sc + wave * 16 * KA_;
    #pragma unroll
    for (int nt = 0; nt < 16; ++nt) {
        int keycol = nt * 16 + (lane & 15);
        const __bf16* krow = &kS[keycol * (HD_ + 8)];
        v8f acc = {};
        #pragma unroll
        for (int s = 0; s < 2; ++s) {
            v16bf bk_;
            #pragma unroll
            for (int vv = 0; vv < 8; ++vv) {
                int kd = s * 32 + ((vv < 4) ? (khalf * 8 + 2 * vv) : (16 + khalf * 8 + 2 * (vv - 4)));
                bk_[2 * vv]     = krow[kd];
                bk_[2 * vv + 1] = krow[kd + 1];
            }
            acc = __builtin_amdgcn_wmma_f32_16x16x32_bf16(
                false, aq[s], false, bk_, (short)0, acc, false, false);
        }
        #pragma unroll
        for (int r = 0; r < 8; ++r)
            scw[(r + 8 * khalf) * KA_ + keycol] = acc[r] * SCALE_;
    }
    __syncthreads();

    // row softmax: exp stored back to sc, 1/sum to rinv
    if (tid < 64) {
        float* row = sc + tid * KA_;
        float m = -3.0e38f;
        for (int j = 0; j < KA_; ++j) m = fmaxf(m, row[j]);
        float ssum = 0.f;
        for (int j = 0; j < KA_; ++j) { float e = __expf(row[j] - m); row[j] = e; ssum += e; }
        rinv[tid] = 1.f / ssum;
    }
    __syncthreads();

    // ctx = softmax(scores) @ V
    v8f cacc[4] = {};
    const float* scrow = sc + (wave * 16 + (lane & 15)) * KA_;
    const float inv = rinv[wave * 16 + (lane & 15)];
    #pragma unroll
    for (int kb = 0; kb < 8; ++kb) {
        v16bf aw;
        #pragma unroll
        for (int vv = 0; vv < 8; ++vv) {
            int kk = kb * 32 + ((vv < 4) ? (khalf * 8 + 2 * vv) : (16 + khalf * 8 + 2 * (vv - 4)));
            aw[2 * vv]     = (__bf16)(scrow[kk]     * inv);
            aw[2 * vv + 1] = (__bf16)(scrow[kk + 1] * inv);
        }
        #pragma unroll
        for (int dt = 0; dt < 4; ++dt) {
            int dcol = dt * 16 + (lane & 15);
            const __bf16* vrow = &vT[dcol * (KA_ + 8)];
            v16bf bv_;
            #pragma unroll
            for (int vv = 0; vv < 8; ++vv) {
                int kk = kb * 32 + ((vv < 4) ? (khalf * 8 + 2 * vv) : (16 + khalf * 8 + 2 * (vv - 4)));
                bv_[2 * vv]     = vrow[kk];
                bv_[2 * vv + 1] = vrow[kk + 1];
            }
            cacc[dt] = __builtin_amdgcn_wmma_f32_16x16x32_bf16(
                false, aw, false, bv_, (short)0, cacc[dt], false, false);
        }
    }

    __bf16* cB = ctx + ((long)b * N_ + qBlock + wave * 16) * D_ + h * HD_;
    #pragma unroll
    for (int dt = 0; dt < 4; ++dt) {
        int col = dt * 16 + (lane & 15);
        #pragma unroll
        for (int r = 0; r < 8; ++r)
            cB[(long)(r + 8 * khalf) * D_ + col] = (__bf16)cacc[dt][r];
    }
}

// ---------------------------------------------------------------- host
extern "C" void kernel_launch(void* const* d_in, const int* in_sizes, int n_in,
                              void* d_out, int out_size, void* d_ws, size_t ws_size,
                              hipStream_t stream) {
    const float* x   = (const float*)d_in[0];
    const float* Wq  = (const float*)d_in[1];
    const float* bq  = (const float*)d_in[2];
    const float* Wk  = (const float*)d_in[3];
    const float* bk  = (const float*)d_in[4];
    const float* Wv  = (const float*)d_in[5];
    const float* bv  = (const float*)d_in[6];
    const float* Wqt = (const float*)d_in[7];
    const float* bqt = (const float*)d_in[8];
    const float* Wo  = (const float*)d_in[9];
    const float* bo  = (const float*)d_in[10];

    char* ws = (char*)d_ws;
    size_t off = 0;
    auto alloc = [&](size_t bytes) -> void* {
        void* p = ws + off; off += (bytes + 255) & ~(size_t)255; return p;
    };
    const size_t XE = (size_t)B_ * N_ * D_;      // 16M elems
    const size_t WE = (size_t)D_ * D_;           // 1M elems
    __bf16* xb   = (__bf16*)alloc(XE * 2);
    __bf16* Wqb  = (__bf16*)alloc(WE * 2);
    __bf16* Wkb  = (__bf16*)alloc(WE * 2);
    __bf16* Wvb  = (__bf16*)alloc(WE * 2);
    __bf16* Wqtb = (__bf16*)alloc(WE * 2);
    __bf16* Wob  = (__bf16*)alloc(WE * 2);
    __bf16* qbuf = (__bf16*)alloc(XE * 2);
    __bf16* kbuf = (__bf16*)alloc((size_t)B_ * KA_ * D_ * 2);
    __bf16* vbuf = (__bf16*)alloc((size_t)B_ * KA_ * D_ * 2);
    __bf16* ctxb = xb;   // x-bf16 is dead after the projections; reuse for ctx

    // 1. downconvert inputs to bf16 (vectorized x4)
    cvt_f32_bf16_k<<<(int)(XE / 1024), 256, 0, stream>>>(x,   xb,   (int)(XE / 4));
    cvt_f32_bf16_k<<<(int)(WE / 1024), 256, 0, stream>>>(Wq,  Wqb,  (int)(WE / 4));
    cvt_f32_bf16_k<<<(int)(WE / 1024), 256, 0, stream>>>(Wk,  Wkb,  (int)(WE / 4));
    cvt_f32_bf16_k<<<(int)(WE / 1024), 256, 0, stream>>>(Wv,  Wvb,  (int)(WE / 4));
    cvt_f32_bf16_k<<<(int)(WE / 1024), 256, 0, stream>>>(Wqt, Wqtb, (int)(WE / 4));
    cvt_f32_bf16_k<<<(int)(WE / 1024), 256, 0, stream>>>(Wo,  Wob,  (int)(WE / 4));

    const long sX = (long)N_ * D_;
    const long sKV = (long)KA_ * D_;

    // 2. projections (batched over grid.z); N tiles = 1024/128 = 8
    gemm_bf16_wmma_k<false><<<dim3(8, 2, B_), 256, 0, stream>>>(    // q anchors (256 rows)
        xb, D_, sX, Wqb, D_, bq, qbuf, D_, sX, D_);
    gemm_bf16_wmma_k<false><<<dim3(8, 30, B_), 256, 0, stream>>>(   // q queries (3840 rows)
        xb + (size_t)KA_ * D_, D_, sX, Wqtb, D_, bqt,
        qbuf + (size_t)KA_ * D_, D_, sX, D_);
    gemm_bf16_wmma_k<false><<<dim3(8, 2, B_), 256, 0, stream>>>(    // k
        xb, D_, sX, Wkb, D_, bk, kbuf, D_, sKV, D_);
    gemm_bf16_wmma_k<false><<<dim3(8, 2, B_), 256, 0, stream>>>(    // v
        xb, D_, sX, Wvb, D_, bv, vbuf, D_, sKV, D_);

    // 3. attention (64 queries per block, all 256 anchor keys in LDS)
    attn_wmma_k<<<dim3(N_ / 64, H_, B_), 128, ATTN_SMEM, stream>>>(qbuf, kbuf, vbuf, ctxb);

    // 4. output projection (f32 out + bias)
    gemm_bf16_wmma_k<true><<<dim3(8, 32, B_), 256, 0, stream>>>(
        ctxb, D_, sX, Wob, D_, bo, d_out, D_, sX, D_);
}
